// SelectiveScaledDotProductGatedAttention_87393994539286
// MI455X (gfx1250) — matrix operand order repaired and checked by reading, not compiled, not run on hardware
//
#include <hip/hip_runtime.h>
#include <hip/hip_bf16.h>
#include <stdint.h>

// Fused gated-sigmoid attention for MI455X (gfx1250, wave32, WMMA).
// attn = sigmoid(mask ? -inf : (q@k^T)/8 * gate) ; out = attn^T @ v
// Memory-bound (~640MB HBM traffic -> ~27us floor @ 23.3TB/s); both matmuls
// run on v_wmma_f32_16x16x32_bf16. bf16 conversion done with v_perm_b32
// (truncation) and sigmoid with v_exp_f32/v_rcp_f32 so VALU issue stays
// under the HBM roofline. LDS rows padded so ds_load_b128 is conflict-free.

typedef __attribute__((ext_vector_type(16))) __bf16 v16bf;
typedef __attribute__((ext_vector_type(8)))  float v8f;

#define BATCH 64
#define SEQ   1024
#define DIM   64
#define KKT   128   // kk columns per workgroup
#define QT    32    // q rows per step (== K of second WMMA)
#define NW    8     // waves per workgroup
// sigmoid(z/8) = 1 / (1 + exp2(z * (-log2(e)/8)))
#define SIGSCL (-0.18033688011112042f)

// padded LDS row strides (ushorts): byte stride 144 -> bank step 36 (16-lane
// spread), byte stride 80 -> bank step 20 (16-lane spread); both keep 16B align.
#define KSTR 72
#define QSTR 72
#define VSTR 40
#define ASTR 40

union V16U { v16bf v; uint4 q[2]; };

__device__ __forceinline__ v16bf lds_pair(const unsigned short* p0,
                                          const unsigned short* p1) {
  V16U t;
  t.q[0] = *(const uint4*)p0;   // ds_load_b128
  t.q[1] = *(const uint4*)p1;   // ds_load_b128
  return t.v;
}
__device__ __forceinline__ v16bf lds_contig(const unsigned short* p) {
  return lds_pair(p, p + 8);
}
// pack two f32 -> two bf16 (truncate) in ONE v_perm_b32
__device__ __forceinline__ unsigned pk2bf(float x, float y) {
  union { float f; unsigned u; } a, b; a.f = x; b.f = y;
  return __builtin_amdgcn_perm(b.u, a.u, 0x07060302u);
}
__device__ __forceinline__ unsigned short f2bf(float f) {  // truncate
  union { float f; unsigned u; } x; x.f = f;
  return (unsigned short)(x.u >> 16);
}
__device__ __forceinline__ v8f wmma_bf16(v16bf a, v16bf b, v8f c) {
  return __builtin_amdgcn_wmma_f32_16x16x32_bf16(false, a, false, b,
                                                 (short)0, c, false, false);
}

__global__ __launch_bounds__(256) void fused_gated_attn(
    const float* __restrict__ q, const float* __restrict__ k,
    const float* __restrict__ v, const unsigned char* __restrict__ mask,
    const float* __restrict__ gate, float* __restrict__ out,
    float* __restrict__ attn)
{
  __shared__ __align__(16) unsigned short ks[KKT * KSTR];    // [kk][d] bf16
  __shared__ __align__(16) unsigned short qs[QT * QSTR];     // [q][d]  bf16
  __shared__ __align__(16) unsigned short vT[DIM * VSTR];    // [d][q]  bf16
  __shared__ __align__(16) unsigned short aT[NW * 16 * ASTR];// per-wave [kk16][q32]

  const int bIdx = blockIdx.x / (SEQ / KKT);
  const int kt   = blockIdx.x % (SEQ / KKT);
  const int kk0  = kt * KKT;
  const int tid  = threadIdx.x;
  const int wave = tid >> 5;
  const int lane = tid & 31;
  const int m    = lane & 15;   // N column / M row selector
  const int hi   = lane >> 4;   // half-wave selector

  // ---- stage K tile (KKT x DIM) as bf16, coalesced float4 reads ----
  {
    const float* kb = k + ((size_t)bIdx * SEQ + kk0) * DIM;
    #pragma unroll
    for (int it = 0; it < (KKT * DIM) / 1024; ++it) {
      const int i = it * 1024 + tid * 4;
      float4 f = *(const float4*)(kb + i);
      uint2 pk;
      pk.x = pk2bf(f.x, f.y);
      pk.y = pk2bf(f.z, f.w);
      *(uint2*)(ks + (i >> 6) * KSTR + (i & 63)) = pk;
    }
  }
  __syncthreads();

  // ---- loop-invariant B operands for q@k^T: column n = ks row wave*16+n ----
  const int kkw = wave * 16 + m;
  v16bf bk0 = lds_pair(&ks[kkw * KSTR +  0 + hi * 8], &ks[kkw * KSTR + 16 + hi * 8]);  // d 0..31
  v16bf bk1 = lds_pair(&ks[kkw * KSTR + 32 + hi * 8], &ks[kkw * KSTR + 48 + hi * 8]);  // d 32..63

  v8f o0 = {}, o1 = {}, o2 = {}, o3 = {};  // out accum: 16kk x 64d

  const int kkg = kk0 + wave * 16 + m;     // this lane's attn column
  unsigned short* aTw = &aT[wave * 16 * ASTR];

  for (int q0 = 0; q0 < SEQ; q0 += QT) {
    __syncthreads();  // previous iteration consumers done
    const float* qb = q + ((size_t)bIdx * SEQ + q0) * DIM;
    const float* vb = v + ((size_t)bIdx * SEQ + q0) * DIM;
    #pragma unroll
    for (int it = 0; it < (QT * DIM) / 1024; ++it) {
      const int i = it * 1024 + tid * 4;
      float4 f = *(const float4*)(qb + i);
      uint2 pk;
      pk.x = pk2bf(f.x, f.y);
      pk.y = pk2bf(f.z, f.w);
      *(uint2*)(qs + (i >> 6) * QSTR + (i & 63)) = pk;
      float4 g = *(const float4*)(vb + i);
      const int qq = i >> 6, d0 = i & 63;  // transpose V into [d][q]
      vT[(d0 + 0) * VSTR + qq] = f2bf(g.x);
      vT[(d0 + 1) * VSTR + qq] = f2bf(g.y);
      vT[(d0 + 2) * VSTR + qq] = f2bf(g.z);
      vT[(d0 + 3) * VSTR + qq] = f2bf(g.w);
    }
    __syncthreads();

    // ---- matmul 1: S(32q x 16kk) = Qblk @ Ktile^T, K=64 in two WMMAs ----
    v8f s0 = {}, s1 = {};
    {
      v16bf a00 = lds_pair(&qs[m * QSTR +  0 + hi * 8], &qs[m * QSTR + 16 + hi * 8]);
      v16bf a01 = lds_pair(&qs[m * QSTR + 32 + hi * 8], &qs[m * QSTR + 48 + hi * 8]);
      s0 = wmma_bf16(a00, bk0, s0);
      s0 = wmma_bf16(a01, bk1, s0);
      const int m2 = m + 16;
      v16bf a10 = lds_pair(&qs[m2 * QSTR +  0 + hi * 8], &qs[m2 * QSTR + 16 + hi * 8]);
      v16bf a11 = lds_pair(&qs[m2 * QSTR + 32 + hi * 8], &qs[m2 * QSTR + 48 + hi * 8]);
      s1 = wmma_bf16(a10, bk0, s1);
      s1 = wmma_bf16(a11, bk1, s1);
    }

    // ---- gate * scale, mask, sigmoid; stream attn out; bf16 into aT ----
    // C layout: lane {n=m, hi}, VGPR r -> row r+8*hi. Each lane writes aT
    // row m at q-offsets {16*half + 8*hi + r}, while the A-operand reload below
    // reads q 16*hi..16*hi+15, half written by the other half-wave (same wave).
    const size_t ebase = ((size_t)bIdx * SEQ + q0) * SEQ + kkg;
    const float* gp = gate + ebase;
    const unsigned char* mp = mask + ebase;
    float* ap = attn + ebase;
    if (q0 + QT < SEQ) {  // prefetch next q-step's gate/mask stream
      __builtin_prefetch(gp + (size_t)(QT + 8 * hi) * SEQ, 0, 0);
      __builtin_prefetch(gp + (size_t)(QT + 16 + 8 * hi) * SEQ, 0, 0);
      __builtin_prefetch(mp + (size_t)(QT + 8 * hi) * SEQ, 0, 0);
      __builtin_prefetch(mp + (size_t)(QT + 16 + 8 * hi) * SEQ, 0, 0);
    }
    #pragma unroll
    for (int half = 0; half < 2; ++half) {
      const v8f S = half ? s1 : s0;
      const int qloc = half * 16 + 8 * hi;
      const float* gph = gp + (size_t)qloc * SEQ;
      const unsigned char* mph = mp + (size_t)qloc * SEQ;
      float* aph = ap + (size_t)qloc * SEQ;
      #pragma unroll
      for (int rp = 0; rp < 4; ++rp) {
        float av[2];
        #pragma unroll
        for (int j = 0; j < 2; ++j) {
          const int r = rp * 2 + j;
          float g = __builtin_nontemporal_load(gph + r * SEQ);  // imm offset
          unsigned char mk = __builtin_nontemporal_load(mph + r * SEQ);
          float t = S[r] * g;
          float e = __builtin_amdgcn_exp2f(t * SIGSCL);          // v_exp_f32
          float a = (mk != 0) ? 0.0f
                              : __builtin_amdgcn_rcpf(1.0f + e); // v_rcp_f32
          __builtin_nontemporal_store(a, aph + r * SEQ);
          av[j] = a;
        }
        *(unsigned*)(aTw + m * ASTR + qloc + rp * 2) = pk2bf(av[0], av[1]);
      }
    }

    // ---- matmul 2: O(16kk x 64d) += attn^T(16x32) @ Vblk(32x64) ----
    v16bf aa = lds_contig(aTw + m * ASTR + hi * 16);
    v16bf bv0 = lds_contig(&vT[( 0 + m) * VSTR + hi * 16]);
    v16bf bv1 = lds_contig(&vT[(16 + m) * VSTR + hi * 16]);
    v16bf bv2 = lds_contig(&vT[(32 + m) * VSTR + hi * 16]);
    v16bf bv3 = lds_contig(&vT[(48 + m) * VSTR + hi * 16]);
    o0 = wmma_bf16(aa, bv0, o0);
    o1 = wmma_bf16(aa, bv1, o1);
    o2 = wmma_bf16(aa, bv2, o2);
    o3 = wmma_bf16(aa, bv3, o3);
  }

  // ---- epilogue: out[b][kk0 + wave*16 + r + 8*hi][t*16 + m] ----
  const int kkb = kk0 + wave * 16 + 8 * hi;
  #pragma unroll
  for (int t = 0; t < 4; ++t) {
    v8f O = (t == 0) ? o0 : (t == 1) ? o1 : (t == 2) ? o2 : o3;
    #pragma unroll
    for (int r = 0; r < 8; ++r) {
      out[((size_t)bIdx * SEQ + kkb + r) * DIM + t * 16 + m] = O[r];
    }
  }
}

extern "C" void kernel_launch(void* const* d_in, const int* in_sizes, int n_in,
                              void* d_out, int out_size, void* d_ws, size_t ws_size,
                              hipStream_t stream) {
  (void)in_sizes; (void)n_in; (void)out_size; (void)d_ws; (void)ws_size;
  const float* q = (const float*)d_in[0];
  const float* k = (const float*)d_in[1];
  const float* v = (const float*)d_in[2];
  const unsigned char* mask = (const unsigned char*)d_in[3];  // jnp bool = 1 byte
  const float* gate = (const float*)d_in[4];
  float* out  = (float*)d_out;                                  // [B,L,D]
  float* attn = out + (size_t)BATCH * SEQ * DIM;                // [B,L,L]
  dim3 grid(BATCH * (SEQ / KKT));  // 512 workgroups
  dim3 block(256);                 // 8 wave32 waves
  fused_gated_attn<<<grid, block, 0, stream>>>(q, k, v, mask, gate, out, attn);
}
